// VQVAEQuantize_85796266705314
// MI455X (gfx1250) — matrix operand order repaired and checked by reading, not compiled, not run on hardware
//
#include <hip/hip_runtime.h>

typedef __attribute__((ext_vector_type(2)))  float        v2f;
typedef __attribute__((ext_vector_type(8)))  float        v8f;
typedef __attribute__((ext_vector_type(16))) __bf16       v16bf;
typedef __attribute__((ext_vector_type(4)))  unsigned int u32x4;
typedef __attribute__((ext_vector_type(4)))  int          i32x4;
typedef __attribute__((ext_vector_type(8)))  int          i32x8;

#define PITCH 65        // odd float pitch -> conflict-free LDS columns
#define ROWB  272       // padded LDS row pitch for bf16 code tiles (bytes)

#if defined(__has_builtin)
#if __has_builtin(__builtin_amdgcn_tensor_load_to_lds)
#define HAVE_TDM 1
#endif
#endif
#ifndef HAVE_TDM
#define HAVE_TDM 0
#endif

// ---------------------------------------------------------------------------
// ws layout: float[0] loss accum; float[64..8256) = 0.5*||w_j||^2;
//            byte 33024.. : bf16 codebook, 8192 rows x 256B ([64 hi][64 lo])
// d_out: [0,2097152) out[B,E,H,W]; [2097152,4194304) z_q_flat;
//        [4194304] latent_loss; [4194305,4227073) z_q_ind
// ---------------------------------------------------------------------------
#define WS_WN 64
#define WS_CB_BYTES 33024

__device__ __forceinline__ unsigned short f2bf_rne(float f) {
  unsigned u = __builtin_bit_cast(unsigned, f);
  unsigned r = u + 0x7FFFu + ((u >> 16) & 1u);
  return (unsigned short)(r >> 16);
}
__device__ __forceinline__ float bf2f(unsigned short h) {
  unsigned u = ((unsigned)h) << 16;
  return __builtin_bit_cast(float, u);
}

// ---- prep: 0.5*||w||^2 + split-bf16 codebook + zero loss accumulator ------
__global__ void vq_prep_kernel(const float* __restrict__ embed,
                               float* __restrict__ ws) {
  int j = blockIdx.x * blockDim.x + threadIdx.x;   // 0..8191
  if (j == 0) ws[0] = 0.0f;
  const float* row = embed + (size_t)j * 64;
  unsigned short* dst =
      (unsigned short*)((char*)ws + WS_CB_BYTES + (size_t)j * 256);
  float s = 0.f;
  #pragma unroll
  for (int e = 0; e < 64; ++e) {
    float x = row[e];
    s += x * x;
    unsigned short h = f2bf_rne(x);
    dst[e] = h;
    dst[64 + e] = f2bf_rne(x - bf2f(h));
  }
  ws[WS_WN + j] = 0.5f * s;
}

#if HAVE_TDM
// 1D TDM copy of one 4KB code tile (1024 dwords) with LDS padding:
// 4 dwords inserted every 64 dwords -> 272B effective row pitch.
__device__ __forceinline__ void tdm_load_tile(const char* gsrc,
                                              unsigned lds_addr) {
  unsigned long long ga = (unsigned long long)(size_t)gsrc;
  u32x4 g0;
  g0.x = 1u;                                   // count=1, user descriptor
  g0.y = lds_addr;                             // LDS byte address
  g0.z = (unsigned)ga;                         // global_addr[31:0]
  g0.w = (unsigned)((ga >> 32) & 0x1FFFFFFu) | (2u << 30);  // [56:32] | type=2
  i32x8 g1;
  // data_size=4B(2), pad_enable, pad_interval=64dw(5), pad_amount=4dw(3)
  g1[0] = (int)((2u << 16) | (1u << 20) | (5u << 22) | (3u << 25));
  g1[1] = 0;                                   // tensor_dim0 low16 = 0 (2^19)
  g1[2] = (int)((524288u >> 16) | (1u << 16)); // tensor_dim0 hi | tensor_dim1=1
  g1[3] = (int)(1024u << 16);                  // tile_dim0 = 1024 dwords
  g1[4] = 0;                                   // tile_dim1/2 unused (1D)
  g1[5] = (int)524288u;                        // tensor_dim0_stride (unused 1D)
  g1[6] = 0;
  g1[7] = 0;
  i32x4 z4 = {0, 0, 0, 0};
#if __clang_major__ >= 23
  i32x8 z8 = {0, 0, 0, 0, 0, 0, 0, 0};
  __builtin_amdgcn_tensor_load_to_lds(g0, g1, z4, z4, z8, 0);
#else
  __builtin_amdgcn_tensor_load_to_lds(g0, g1, z4, z4, 0);
#endif
}
#endif

__device__ __forceinline__ void stage_tile_manual(char* dst, const char* src,
                                                  int tid) {
  const unsigned* s = (const unsigned*)src;
  #pragma unroll
  for (int i = 0; i < 8; ++i) {
    int g = i * 128 + tid;                     // dword 0..1023
    *(unsigned*)(dst + (g >> 6) * ROWB + (g & 63) * 4) = s[g];
  }
}

union A16 { v16bf v; unsigned short s[16]; };
union B16 { v16bf v; u32x4 q[2]; };

__global__ __launch_bounds__(128, 1) void vq_fused_kernel(
    const float* __restrict__ z, const float* __restrict__ pw,
    const float* __restrict__ pb, const float* __restrict__ embed,
    float* __restrict__ out, float* __restrict__ ws) {
  __shared__ float pwT[128 * PITCH];                 // proj_w as [c][e]
  __shared__ float zE[64 * PITCH];                   // z_e tile [row][e]
  __shared__ __align__(16) char ebuf[2][16 * ROWB];  // double-buffered codes
  __shared__ int idxLds[64];

  const int tid  = threadIdx.x;
  const int wave = tid >> 5;
  const int lane = tid & 31;
  const int hi   = lane >> 4;
  const int lm   = lane & 15;
  const int m0   = wave * 16;

  const int n0   = blockIdx.x * 64;
  const int bImg = n0 >> 10;
  const int p0   = n0 & 1023;

  const char* cb = (const char*)ws + WS_CB_BYTES;

#if HAVE_TDM
  // kick DMA of tile 0 immediately; it overlaps the whole projection phase
  if (wave == 0)
    tdm_load_tile(cb, (unsigned)(size_t)&ebuf[0][0]);
#endif

  // ---- stage proj_w [64,128] into LDS transposed to [c][e] ----
  #pragma unroll
  for (int i = 0; i < 64; ++i) {
    int g = i * 128 + tid;
    pwT[(g & 127) * PITCH + (g >> 7)] = pw[g];
  }

  // ---- A operand (z) for projection, straight from global [B,C,H,W] ----
  const float* zb = z + (size_t)bImg * 128 * 1024 + (size_t)(p0 + m0 + lm);
  v2f aP[32];
  #pragma unroll
  for (int s = 0; s < 32; ++s) {
    int c0 = 4 * s + 2 * hi;
    aP[s].x = zb[(size_t)c0 * 1024];
    aP[s].y = zb[(size_t)(c0 + 1) * 1024];
  }
  __syncthreads();

  // ---- projection GEMM (f32 WMMA 16x16x4, K=128): zE = z*pw^T + pb ----
  #pragma unroll
  for (int ct = 0; ct < 4; ++ct) {
    int e0 = ct * 16;
    v8f acc = {};
    #pragma unroll
    for (int s = 0; s < 32; ++s) {
      int k0 = 4 * s + 2 * hi;
      v2f bv;
      bv.x = pwT[k0 * PITCH + e0 + lm];
      bv.y = pwT[(k0 + 1) * PITCH + e0 + lm];
      acc = __builtin_amdgcn_wmma_f32_16x16x4_f32(false, aP[s], false, bv,
                                                  (short)0, acc, false, false);
    }
    float bias = pb[e0 + lm];
    #pragma unroll
    for (int v = 0; v < 8; ++v)
      zE[(m0 + v + 8 * hi) * PITCH + e0 + lm] = acc[v] + bias;
  }
  __syncthreads();

  // ---- build split-bf16 A operands for distance GEMM (held in VGPRs) ----
  // 16-bit A k order used here (contiguous, same as B): elem i -> K=16*hi+i
  A16 aH[2], aL[2];
  #pragma unroll
  for (int k = 0; k < 2; ++k) {
    #pragma unroll
    for (int i = 0; i < 16; ++i) {
      int e = 32 * k + 16 * hi + i;
      float f = zE[(m0 + lm) * PITCH + e];
      unsigned short h = f2bf_rne(f);
      aH[k].s[i] = h;
      aL[k].s[i] = f2bf_rne(f - bf2f(h));
    }
  }

  float rmax[8]; int ridx[8];
  #pragma unroll
  for (int v = 0; v < 8; ++v) { rmax[v] = -3.4e38f; ridx[v] = 0; }

#if HAVE_TDM
  if (wave == 0) __builtin_amdgcn_s_wait_tensorcnt(0);
#else
  stage_tile_manual(ebuf[0], cb, tid);
#endif
  __syncthreads();

  const float* wn = ws + WS_WN;
  // ---- stream all 8192 codes; argmax(x.w - 0.5||w||^2) via 3-term bf16 ----
  for (int jt = 0; jt < 512; ++jt) {
    const int p = jt & 1;
#if HAVE_TDM
    if (wave == 0 && jt + 1 < 512)
      tdm_load_tile(cb + (size_t)(jt + 1) * 4096,
                    (unsigned)(size_t)&ebuf[1 - p][0]);
#endif
    const char* base = ebuf[p] + lm * ROWB;
    const int o = 32 * hi;     // K-half byte offset within 64B chunk section
    B16 bh0, bh1, bl0, bl1;
    bh0.q[0] = *(const u32x4*)(base + o);
    bh0.q[1] = *(const u32x4*)(base + o + 16);
    bh1.q[0] = *(const u32x4*)(base + 64 + o);
    bh1.q[1] = *(const u32x4*)(base + 64 + o + 16);
    bl0.q[0] = *(const u32x4*)(base + 128 + o);
    bl0.q[1] = *(const u32x4*)(base + 128 + o + 16);
    bl1.q[0] = *(const u32x4*)(base + 192 + o);
    bl1.q[1] = *(const u32x4*)(base + 192 + o + 16);

    v8f acc = {};
    acc = __builtin_amdgcn_wmma_f32_16x16x32_bf16(false, aH[0].v, false, bh0.v,
                                                  (short)0, acc, false, false);
    acc = __builtin_amdgcn_wmma_f32_16x16x32_bf16(false, aL[0].v, false, bh0.v,
                                                  (short)0, acc, false, false);
    acc = __builtin_amdgcn_wmma_f32_16x16x32_bf16(false, aH[0].v, false, bl0.v,
                                                  (short)0, acc, false, false);
    acc = __builtin_amdgcn_wmma_f32_16x16x32_bf16(false, aH[1].v, false, bh1.v,
                                                  (short)0, acc, false, false);
    acc = __builtin_amdgcn_wmma_f32_16x16x32_bf16(false, aL[1].v, false, bh1.v,
                                                  (short)0, acc, false, false);
    acc = __builtin_amdgcn_wmma_f32_16x16x32_bf16(false, aH[1].v, false, bl1.v,
                                                  (short)0, acc, false, false);

    float wnv = wn[jt * 16 + lm];
    int jcol = jt * 16 + lm;
    #pragma unroll
    for (int v = 0; v < 8; ++v) {
      float sv = acc[v] - wnv;
      if (sv > rmax[v]) { rmax[v] = sv; ridx[v] = jcol; }  // keeps lowest j
    }

#if HAVE_TDM
    if (wave == 0) __builtin_amdgcn_s_wait_tensorcnt(0);
    __syncthreads();
#else
    __syncthreads();
    if (jt + 1 < 512)
      stage_tile_manual(ebuf[1 - p], cb + (size_t)(jt + 1) * 4096, tid);
    __syncthreads();
#endif
  }

  // ---- argmax across the 16 code-lanes of each half-wave ----
  #pragma unroll
  for (int v = 0; v < 8; ++v) {
    float mv = rmax[v]; int iv = ridx[v];
    #pragma unroll
    for (int off = 1; off < 16; off <<= 1) {
      float om = __shfl_xor(mv, off, 32);
      int   oi = __shfl_xor(iv, off, 32);
      if (om > mv || (om == mv && oi < iv)) { mv = om; iv = oi; }
    }
    if (lm == 0) idxLds[m0 + v + 8 * hi] = iv;
  }
  __syncthreads();

  // ---- gather winners (exact fp32 codebook), outputs, loss ----
  const int r    = tid >> 1;
  const int half = tid & 1;
  const int n    = n0 + r;
  const int idx  = idxLds[r];
  const float* er = embed + (size_t)idx * 64;
  float lsum = 0.f;
  #pragma unroll
  for (int i = 0; i < 32; ++i) {
    int e = half * 32 + i;
    float zq = er[e];
    float ze = zE[r * PITCH + e];
    float d = zq - ze;
    lsum += d * d;
    out[2097152 + (size_t)n * 64 + e] = zq;                                  // z_q_flat
    out[(size_t)bImg * 65536 + (size_t)e * 1024 + (size_t)(n & 1023)] = zq;  // out
  }
  if (half == 0)
    out[4194305 + n] = (float)idx;   // z_q_ind

  #pragma unroll
  for (int off = 1; off < 32; off <<= 1)
    lsum += __shfl_xor(lsum, off, 32);
  if (lane == 0) atomicAdd(&ws[0], lsum);
}

__global__ void vq_finalize_kernel(const float* __restrict__ ws,
                                   float* __restrict__ out) {
  out[4194304] = 12.5f * ws[0] * (1.0f / 2097152.0f);
}

extern "C" void kernel_launch(void* const* d_in, const int* in_sizes, int n_in,
                              void* d_out, int out_size, void* d_ws, size_t ws_size,
                              hipStream_t stream) {
  const float* z     = (const float*)d_in[0];
  const float* pw    = (const float*)d_in[1];
  const float* pb    = (const float*)d_in[2];
  const float* embed = (const float*)d_in[3];
  float* out = (float*)d_out;
  float* ws  = (float*)d_ws;

  vq_prep_kernel<<<64, 128, 0, stream>>>(embed, ws);
  vq_fused_kernel<<<512, 128, 0, stream>>>(z, pw, pb, embed, out, ws);
  vq_finalize_kernel<<<1, 1, 0, stream>>>(ws, out);
}
